// TransformerBlock_83708912599171
// MI455X (gfx1250) — matrix operand order repaired and checked by reading, not compiled, not run on hardware
//
#include <hip/hip_runtime.h>
#include <hip/hip_bf16.h>

// ---------------------------------------------------------------------------
// Transformer block forward for MI455X (gfx1250), wave32 + WMMA f16->f32,
// GEMM tiles staged into LDS by the Tensor Data Mover (TDM) with hardware
// row padding, double-buffered via TENSORcnt + workgroup barriers.
// DIM=512, HEADS=8, HEAD_DIM=64, HIDDEN=2048, B=2, N=4096.
// ---------------------------------------------------------------------------

typedef __attribute__((ext_vector_type(16))) _Float16 v16h;
typedef __attribute__((ext_vector_type(8)))  _Float16 v8h;
typedef __attribute__((ext_vector_type(8)))  float    v8f;

typedef __attribute__((ext_vector_type(4))) unsigned int tdm_v4u;
typedef __attribute__((ext_vector_type(8))) int          tdm_v8i;
typedef __attribute__((ext_vector_type(4))) int          tdm_v4i;

#define DIM     512
#define HEADS   8
#define HDIM    64
#define HIDDEN  2048
#define BATCH   2
#define SEQ     4096
#define ROWS    (BATCH * SEQ)   // 8192

#if __has_builtin(__builtin_amdgcn_tensor_load_to_lds) && \
    __has_builtin(__builtin_amdgcn_s_wait_tensorcnt)
#define USE_TDM 1
#else
#define USE_TDM 0
#endif

__device__ __forceinline__ v16h make_frag(v8h lo, v8h hi) {
  return __builtin_shufflevector(lo, hi, 0,1,2,3,4,5,6,7,8,9,10,11,12,13,14,15);
}

__device__ __forceinline__ v8f wmma_f16(v16h a, v16h b, v8f c) {
  // 8 args: (neg_a, A, neg_b, B, c_mod, C, reuse_a, reuse_b)
  return __builtin_amdgcn_wmma_f32_16x16x32_f16(false, a, false, b, (short)0, c,
                                                false, false);
}

#if USE_TDM
// ---------------------------------------------------------------------------
// TDM: load a 2D f16 tile [tile_rows x 32] from a row-major [tensor_rows x
// tensor_k] tensor into LDS, padding each 64B row with 16B so LDS rows land
// at a 40-half (80B) stride — identical to the fragment readers' layout.
// D# layout per CDNA5 ISA section 8.3/8.4. 2D tile => groups 2/3 all zero
// (tile_dim2/3/4 = 0 = unused).
// ---------------------------------------------------------------------------
__device__ __forceinline__ void tdm_load_tile(const _Float16* gsrc,
                                              unsigned lds_off,
                                              int tile_rows, int tensor_rows,
                                              int tensor_k) {
  unsigned long long ga = (unsigned long long)(__UINTPTR_TYPE__)gsrc;
  tdm_v4u g0;
  g0[0] = 1u;                                      // count=1, user descriptor
  g0[1] = lds_off;                                 // lds_addr (bytes)
  g0[2] = (unsigned)(ga & 0xffffffffu);            // global_addr[31:0]
  g0[3] = (unsigned)((ga >> 32) & 0x01ffffffu)     // global_addr[56:32]
        | (2u << 30);                              // type = 2 ("image")
  tdm_v8i g1;
  // dword0: workgroup_mask=0 | data_size=1 (2B) | pad_enable
  //         | pad_interval=3 (16 DWORDs = one 64B row) | pad_amount=3 (4 DWORDs)
  g1[0] = (1 << 16) | (1 << 20) | (3 << 22) | (3 << 25);
  // dword1: atomic_barrier_addr=0 [15:0]; tensor_dim0[15:0] in [31:16]
  g1[1] = (int)(((unsigned)tensor_k & 0xffffu) << 16);
  // dword2: tensor_dim0[31:16] | tensor_dim1[15:0] << 16
  g1[2] = (int)(((unsigned)tensor_k >> 16) |
                (((unsigned)tensor_rows & 0xffffu) << 16));
  // dword3: tensor_dim1[31:16] | tile_dim0 << 16
  g1[3] = (int)(((unsigned)tensor_rows >> 16) | (32u << 16));
  // dword4: tile_dim1 | tile_dim2(=0) << 16
  g1[4] = (int)((unsigned)tile_rows & 0xffffu);
  // dword5/6/7: tensor_dim0_stride (48b) | tensor_dim1_stride (48b, unused 2D)
  g1[5] = (int)(unsigned)tensor_k;
  g1[6] = 0;
  g1[7] = 0;
  tdm_v4i z4 = {0, 0, 0, 0};
#if __clang_major__ >= 23
  tdm_v8i z8 = {0, 0, 0, 0, 0, 0, 0, 0};
  __builtin_amdgcn_tensor_load_to_lds(g0, g1, z4, z4, z8, 0);
#else
  __builtin_amdgcn_tensor_load_to_lds(g0, g1, z4, z4, 0);
#endif
}
#endif  // USE_TDM

// ---------------------------------------------------------------------------
// Weight prep: f32 [K][N] row-major  ->  f16 [N][K] row-major (K contiguous).
// ---------------------------------------------------------------------------
__global__ __launch_bounds__(256)
void cast_transpose(const float* __restrict__ src, _Float16* __restrict__ dst,
                    int K, int N) {
  int idx = blockIdx.x * 256 + threadIdx.x;
  if (idx >= K * N) return;
  int k = idx / N, n = idx - k * N;
  dst[(size_t)n * K + k] = (_Float16)src[idx];
}

// ---------------------------------------------------------------------------
// LayerNorm over C=512 + cast to f16.  One block (256 thr) per row.
// ---------------------------------------------------------------------------
__global__ __launch_bounds__(256)
void ln_cast(const float* __restrict__ x, const float* __restrict__ g,
             const float* __restrict__ bb, _Float16* __restrict__ y) {
  __shared__ float red[256];
  const int row = blockIdx.x, t = threadIdx.x;
  const float* xr = x + (size_t)row * DIM;
  float e0 = xr[t], e1 = xr[t + 256];
  red[t] = e0 + e1;
  __syncthreads();
  #pragma unroll
  for (int d = 128; d > 0; d >>= 1) {
    if (t < d) red[t] += red[t + d];
    __syncthreads();
  }
  float mu = red[0] * (1.0f / DIM);
  __syncthreads();
  float d0 = e0 - mu, d1 = e1 - mu;
  red[t] = d0 * d0 + d1 * d1;
  __syncthreads();
  #pragma unroll
  for (int d = 128; d > 0; d >>= 1) {
    if (t < d) red[t] += red[t + d];
    __syncthreads();
  }
  float rstd = rsqrtf(red[0] * (1.0f / DIM) + 1e-5f);
  y[(size_t)row * DIM + t]       = (_Float16)(d0 * rstd * g[t]       + bb[t]);
  y[(size_t)row * DIM + t + 256] = (_Float16)(d1 * rstd * g[t + 256] + bb[t + 256]);
}

// ---------------------------------------------------------------------------
// WMMA GEMM: C[M,N] = A[M,K] (f16, K-contig) * Bt[N,K]^T (f16, K-contig)
// Block tile 128x64, BK=32, 8 waves, each wave 32x32 (2x2 WMMA tiles).
// Tiles staged into double-buffered LDS by TDM (wave 0 issues, TENSORcnt +
// barrier publishes), with TDM hardware padding to a 40-half row stride.
// Epilogues: QKV head split / bias+GELU->f16 / bias+residual->f32.
// ---------------------------------------------------------------------------
enum { EPI_QKV = 0, EPI_GELU = 1, EPI_RES = 2 };

template <int EPI>
__global__ __launch_bounds__(256)
void gemm_wmma(const _Float16* __restrict__ A, const _Float16* __restrict__ Bt,
               const float* __restrict__ bias, const float* __restrict__ res,
               float* __restrict__ outF, _Float16* __restrict__ outH,
               _Float16* __restrict__ qb, _Float16* __restrict__ kb,
               _Float16* __restrict__ vtb, int M, int N, int K) {
  constexpr int BM = 128, BN = 64, BK = 32, PAD = 40;  // 40-half rows
  __shared__ __align__(16) _Float16 As[2][BM * PAD];
  __shared__ __align__(16) _Float16 Bs[2][BN * PAD];

  const int tid = threadIdx.x;
  const int lane = tid & 31, w = tid >> 5;
  const int hl = lane >> 4, l16 = lane & 15;
  const int wm = w >> 1, wn = w & 1;          // 4 x 2 wave grid
  const int m0 = blockIdx.x * BM;
  const int n0 = blockIdx.y * BN;
  const int nsteps = K / BK;

  v8f acc[2][2] = {};

#if USE_TDM
  if (w == 0) {  // prologue: DMA tile 0 into buffer 0
    tdm_load_tile(&A[(size_t)m0 * K],  (unsigned)(__UINTPTR_TYPE__)&As[0][0],
                  BM, M, K);
    tdm_load_tile(&Bt[(size_t)n0 * K], (unsigned)(__UINTPTR_TYPE__)&Bs[0][0],
                  BN, N, K);
  }
#endif

  for (int step = 0; step < nsteps; ++step) {
    const int buf = step & 1;
#if USE_TDM
    if (w == 0) __builtin_amdgcn_s_wait_tensorcnt(0);
    __syncthreads();                       // publish DMA'd LDS to all waves
    if (w == 0 && step + 1 < nsteps) {     // DMA next tile into other buffer
      const int k1 = (step + 1) * BK;
      tdm_load_tile(&A[(size_t)m0 * K + k1],
                    (unsigned)(__UINTPTR_TYPE__)&As[buf ^ 1][0], BM, M, K);
      tdm_load_tile(&Bt[(size_t)n0 * K + k1],
                    (unsigned)(__UINTPTR_TYPE__)&Bs[buf ^ 1][0], BN, N, K);
    }
#else
    {  // fallback: manual cooperative copy
      const int k0 = step * BK;
      #pragma unroll
      for (int i = 0; i < 2; ++i) {
        int c = tid + i * 256;
        int row = c >> 2, off = (c & 3) * 8;
        *(v8h*)&As[buf][row * PAD + off] =
            *(const v8h*)&A[(size_t)(m0 + row) * K + k0 + off];
      }
      {
        int row = tid >> 2, off = (tid & 3) * 8;
        *(v8h*)&Bs[buf][row * PAD + off] =
            *(const v8h*)&Bt[(size_t)(n0 + row) * K + k0 + off];
      }
      __syncthreads();
    }
#endif

    v16h af[2], bf[2];
    #pragma unroll
    for (int mt = 0; mt < 2; ++mt) {
      int r = wm * 32 + mt * 16 + l16;
      v8h lo = *(const v8h*)&As[buf][r * PAD + 8 * hl];       // K 0..7 / 8..15
      v8h hi = *(const v8h*)&As[buf][r * PAD + 16 + 8 * hl];  // K 16..23 / 24..31
      af[mt] = make_frag(lo, hi);
    }
    #pragma unroll
    for (int nt = 0; nt < 2; ++nt) {
      int r = wn * 32 + nt * 16 + l16;
      v8h lo = *(const v8h*)&Bs[buf][r * PAD + 8 * hl];       // K 0..7 / 8..15
      v8h hi = *(const v8h*)&Bs[buf][r * PAD + 16 + 8 * hl];  // K 16..23 / 24..31
      bf[nt] = make_frag(lo, hi);
    }
    #pragma unroll
    for (int mt = 0; mt < 2; ++mt)
      #pragma unroll
      for (int nt = 0; nt < 2; ++nt)
        acc[mt][nt] = wmma_f16(af[mt], bf[nt], acc[mt][nt]);
    __syncthreads();                       // all reads of buf done
  }

  // epilogue: C element r lives at row 8*hl + r, col l16 within each 16x16 tile
  #pragma unroll
  for (int mt = 0; mt < 2; ++mt) {
    #pragma unroll
    for (int nt = 0; nt < 2; ++nt) {
      #pragma unroll
      for (int r = 0; r < 8; ++r) {
        int gm = m0 + wm * 32 + mt * 16 + 8 * hl + r;
        int gn = n0 + wn * 32 + nt * 16 + l16;
        float v = acc[mt][nt][r] + bias[gn];
        if constexpr (EPI == EPI_RES) {
          outF[(size_t)gm * N + gn] = v + res[(size_t)gm * N + gn];
        } else if constexpr (EPI == EPI_GELU) {
          float ge = 0.5f * v * (1.0f + erff(v * 0.70710678f));
          outH[(size_t)gm * N + gn] = (_Float16)ge;
        } else {  // EPI_QKV: split into Q [bh][n][d], K [bh][n][d], Vt [bh][d][n]
          int bq = gm >> 12, nseq = gm & (SEQ - 1);
          int which = gn >> 9, c2 = gn & (DIM - 1);
          int hh = c2 >> 6, dd = c2 & (HDIM - 1);
          size_t head = (size_t)(bq * HEADS + hh);
          if (which == 0)
            qb[(head * SEQ + nseq) * HDIM + dd] = (_Float16)v;
          else if (which == 1)
            kb[(head * SEQ + nseq) * HDIM + dd] = (_Float16)v;
          else
            vtb[(head * HDIM + dd) * SEQ + nseq] = (_Float16)v;
        }
      }
    }
  }
}

// ---------------------------------------------------------------------------
// Flash attention: one wave = 16 query rows x full head (d=64).
// Block = 4 waves (4 query tiles). Online softmax via wave32 __shfl_xor.
// S = Q*K^T (2 WMMA per 16x16 tile), O += P*V (4 WMMA per key block).
// K/V fragments stream straight from global (L2-resident, 192MB) into WMMA.
// ---------------------------------------------------------------------------
__global__ __launch_bounds__(128)
void flash_attn(const _Float16* __restrict__ Q, const _Float16* __restrict__ Kk,
                const _Float16* __restrict__ Vt, _Float16* __restrict__ O) {
  constexpr int PAD = 40;
  __shared__ __align__(16) _Float16 p_lds[4][16 * PAD];

  const int tid = threadIdx.x, lane = tid & 31, w = tid >> 5;
  const int hl = lane >> 4, l16 = lane & 15;
  int bid = blockIdx.x;               // BATCH*HEADS*(SEQ/64) = 1024
  const int qc = bid & 63; bid >>= 6;
  const int h = bid & (HEADS - 1);
  const int b = bid >> 3;
  const size_t head = (size_t)(b * HEADS + h);
  const _Float16* qp = Q  + head * SEQ * HDIM;
  const _Float16* kp = Kk + head * SEQ * HDIM;
  const _Float16* vp = Vt + head * HDIM * SEQ;
  const int q0 = qc * 64 + w * 16;

  v16h qf[2];
  {
    const size_t row = (size_t)(q0 + l16) * HDIM;
    #pragma unroll
    for (int kk = 0; kk < 2; ++kk) {
      v8h lo = *(const v8h*)&qp[row + kk * 32 + 8 * hl];
      v8h hi = *(const v8h*)&qp[row + kk * 32 + 16 + 8 * hl];
      qf[kk] = make_frag(lo, hi);
    }
  }

  float m_i[8], l_i[8];
  #pragma unroll
  for (int r = 0; r < 8; ++r) { m_i[r] = -1e30f; l_i[r] = 0.0f; }
  v8f o[4] = {};
  const float scale = 0.125f;  // 1/sqrt(64)

  for (int kb = 0; kb < SEQ; kb += 32) {
    // ---- S(16x32) = Q(16x64) x K^T ----
    v8f s[2] = {};
    #pragma unroll
    for (int j = 0; j < 2; ++j) {
      const size_t krow = (size_t)(kb + j * 16 + l16) * HDIM;
      #pragma unroll
      for (int kk = 0; kk < 2; ++kk) {
        v8h lo = *(const v8h*)&kp[krow + kk * 32 + 8 * hl];
        v8h hi = *(const v8h*)&kp[krow + kk * 32 + 16 + 8 * hl];
        s[j] = wmma_f16(qf[kk], make_frag(lo, hi), s[j]);
      }
    }
    // ---- online softmax over the 32 new columns ----
    #pragma unroll
    for (int r = 0; r < 8; ++r) {
      float s0 = s[0][r] * scale, s1 = s[1][r] * scale;
      float mx = fmaxf(s0, s1);
      #pragma unroll
      for (int d = 1; d < 16; d <<= 1) mx = fmaxf(mx, __shfl_xor(mx, d, 32));
      float mnew = fmaxf(m_i[r], mx);
      float p0 = __expf(s0 - mnew), p1 = __expf(s1 - mnew);
      float sum = p0 + p1;
      #pragma unroll
      for (int d = 1; d < 16; d <<= 1) sum += __shfl_xor(sum, d, 32);
      float alpha = __expf(m_i[r] - mnew);
      l_i[r] = l_i[r] * alpha + sum;
      m_i[r] = mnew;
      #pragma unroll
      for (int t = 0; t < 4; ++t) o[t][r] *= alpha;
      const int prow = 8 * hl + r;
      p_lds[w][prow * PAD + l16]      = (_Float16)p0;
      p_lds[w][prow * PAD + 16 + l16] = (_Float16)p1;
    }
    __syncthreads();
    // ---- O(16x64) += P(16x32) x V(32x64) ----
    v16h pf;
    {
      v8h lo = *(const v8h*)&p_lds[w][l16 * PAD + 8 * hl];
      v8h hi = *(const v8h*)&p_lds[w][l16 * PAD + 16 + 8 * hl];
      pf = make_frag(lo, hi);
    }
    #pragma unroll
    for (int t = 0; t < 4; ++t) {
      const size_t vrow = (size_t)(t * 16 + l16) * SEQ + kb;
      v8h lo = *(const v8h*)&vp[vrow + 8 * hl];
      v8h hi = *(const v8h*)&vp[vrow + 16 + 8 * hl];
      o[t] = wmma_f16(pf, make_frag(lo, hi), o[t]);
    }
    __syncthreads();
  }

  // finalize: O /= l, write f16 into [b][n][C] with C = h*64 + col
  #pragma unroll
  for (int r = 0; r < 8; ++r) {
    const float inv = 1.0f / l_i[r];
    const int row = q0 + 8 * hl + r;
    const size_t off = ((size_t)b * SEQ + row) * DIM + h * HDIM;
    #pragma unroll
    for (int t = 0; t < 4; ++t)
      O[off + t * 16 + l16] = (_Float16)(o[t][r] * inv);
  }
}

// ---------------------------------------------------------------------------
// Host-side orchestration.
// ---------------------------------------------------------------------------
extern "C" void kernel_launch(void* const* d_in, const int* in_sizes, int n_in,
                              void* d_out, int out_size, void* d_ws, size_t ws_size,
                              hipStream_t stream) {
  const float* x      = (const float*)d_in[0];
  const float* ln1_g  = (const float*)d_in[1];
  const float* ln1_b  = (const float*)d_in[2];
  const float* w_qkv  = (const float*)d_in[3];
  const float* b_qkv  = (const float*)d_in[4];
  const float* w_proj = (const float*)d_in[5];
  const float* b_proj = (const float*)d_in[6];
  const float* ln2_g  = (const float*)d_in[7];
  const float* ln2_b  = (const float*)d_in[8];
  const float* w_fc1  = (const float*)d_in[9];
  const float* b_fc1  = (const float*)d_in[10];
  const float* w_fc2  = (const float*)d_in[11];
  const float* b_fc2  = (const float*)d_in[12];
  float* out = (float*)d_out;

  char* ws = (char*)d_ws;
  size_t off = 0;
  auto alloc = [&](size_t bytes) -> char* {
    char* p = ws + off;
    off += (bytes + 255) & ~(size_t)255;
    return p;
  };
  _Float16* y1     = (_Float16*)alloc((size_t)ROWS * DIM * 2);
  _Float16* wqkvT  = (_Float16*)alloc((size_t)3 * DIM * DIM * 2);
  _Float16* wprojT = (_Float16*)alloc((size_t)DIM * DIM * 2);
  _Float16* wfc1T  = (_Float16*)alloc((size_t)HIDDEN * DIM * 2);
  _Float16* wfc2T  = (_Float16*)alloc((size_t)DIM * HIDDEN * 2);
  _Float16* qbuf   = (_Float16*)alloc((size_t)ROWS * DIM * 2);
  _Float16* kbuf   = (_Float16*)alloc((size_t)ROWS * DIM * 2);
  _Float16* vtbuf  = (_Float16*)alloc((size_t)ROWS * DIM * 2);
  _Float16* attn_o = (_Float16*)alloc((size_t)ROWS * DIM * 2);
  float*    x1     = (float*)   alloc((size_t)ROWS * DIM * 4);
  _Float16* y2     = (_Float16*)alloc((size_t)ROWS * DIM * 2);
  _Float16* hmid   = (_Float16*)alloc((size_t)ROWS * HIDDEN * 2);

  // --- weight prep (f32 -> f16, transposed so K is contiguous) ---
  cast_transpose<<<(DIM * 3 * DIM + 255) / 256, 256, 0, stream>>>(w_qkv, wqkvT, DIM, 3 * DIM);
  cast_transpose<<<(DIM * DIM + 255) / 256, 256, 0, stream>>>(w_proj, wprojT, DIM, DIM);
  cast_transpose<<<(DIM * HIDDEN + 255) / 256, 256, 0, stream>>>(w_fc1, wfc1T, DIM, HIDDEN);
  cast_transpose<<<(HIDDEN * DIM + 255) / 256, 256, 0, stream>>>(w_fc2, wfc2T, HIDDEN, DIM);

  // --- LN1 ---
  ln_cast<<<ROWS, 256, 0, stream>>>(x, ln1_g, ln1_b, y1);

  // --- QKV gemm: [8192,512] x [512,1536] -> Q/K/Vt f16 ---
  gemm_wmma<EPI_QKV><<<dim3(ROWS / 128, (3 * DIM) / 64), 256, 0, stream>>>(
      y1, wqkvT, b_qkv, nullptr, nullptr, nullptr, qbuf, kbuf, vtbuf,
      ROWS, 3 * DIM, DIM);

  // --- flash attention ---
  flash_attn<<<BATCH * HEADS * (SEQ / 64), 128, 0, stream>>>(qbuf, kbuf, vtbuf, attn_o);

  // --- proj gemm + residual: x1 = x + attn_o @ w_proj + b_proj ---
  gemm_wmma<EPI_RES><<<dim3(ROWS / 128, DIM / 64), 256, 0, stream>>>(
      attn_o, wprojT, b_proj, x, x1, nullptr, nullptr, nullptr, nullptr,
      ROWS, DIM, DIM);

  // --- LN2 ---
  ln_cast<<<ROWS, 256, 0, stream>>>(x1, ln2_g, ln2_b, y2);

  // --- FC1 + GELU ---
  gemm_wmma<EPI_GELU><<<dim3(ROWS / 128, HIDDEN / 64), 256, 0, stream>>>(
      y2, wfc1T, b_fc1, nullptr, nullptr, hmid, nullptr, nullptr, nullptr,
      ROWS, HIDDEN, DIM);

  // --- FC2 + residual -> out ---
  gemm_wmma<EPI_RES><<<dim3(ROWS / 128, DIM / 64), 256, 0, stream>>>(
      hmid, wfc2T, b_fc2, x1, out, nullptr, nullptr, nullptr, nullptr,
      ROWS, DIM, HIDDEN);

  (void)in_sizes; (void)n_in; (void)out_size; (void)ws_size;
}
